// SNN_58841051955724
// MI455X (gfx1250) — matrix operand order repaired and checked by reading, not compiled, not run on hardware
//
#include <hip/hip_runtime.h>
#include <stdint.h>

// ---------------------------------------------------------------------------
// CDNA5 (gfx1250) SNN forward: 3x (GEMM -> LIF).
// Layer 1 (512 x 60000 x 4096 = 252 GFLOP) dominates: bf16 WMMA, f32 accum,
// f32->bf16 via v_perm_b32 pack. Layer 2 A-matrix is exact bf16 spikes and is
// streamed global->LDS with the Tensor Data Mover (TENSOR_LOAD_TO_LDS +
// s_wait_tensorcnt), double buffered.
// ---------------------------------------------------------------------------

typedef __attribute__((ext_vector_type(16))) __bf16 v16bf;
typedef __attribute__((ext_vector_type(8)))  float  v8f;
typedef __attribute__((ext_vector_type(4)))  unsigned int u32x4;
typedef __attribute__((ext_vector_type(8)))  int          i32x8;
typedef __attribute__((ext_vector_type(4)))  int          i32x4;

struct Frag8 { uint4 a, b; };

// f32 -> bf16 round-to-nearest (+0x8000), two values packed with one v_perm_b32
__device__ __forceinline__ unsigned int pack2_bf16(float lo, float hi) {
  unsigned int ul = __float_as_uint(lo) + 0x8000u;
  unsigned int uh = __float_as_uint(hi) + 0x8000u;
  // result bytes {uh[3],uh[2],ul[3],ul[2]}  -> (hi16(uh) << 16) | hi16(ul)
  return __builtin_amdgcn_perm(uh, ul, 0x07060302u);
}

// Each thread owns (row = tid>>1, K-half = (tid&1)*16) of a 128x32 tile.
__device__ __forceinline__ Frag8 fetch_tile_f32(const float* __restrict__ src,
                                                size_t ld, int tid) {
  const int row = tid >> 1;
  const int ko  = (tid & 1) * 16;
  const float4* p = (const float4*)(src + (size_t)row * ld + ko);
  float4 f0 = p[0], f1 = p[1], f2 = p[2], f3 = p[3];
  Frag8 r;
  r.a.x = pack2_bf16(f0.x, f0.y); r.a.y = pack2_bf16(f0.z, f0.w);
  r.a.z = pack2_bf16(f1.x, f1.y); r.a.w = pack2_bf16(f1.z, f1.w);
  r.b.x = pack2_bf16(f2.x, f2.y); r.b.y = pack2_bf16(f2.z, f2.w);
  r.b.z = pack2_bf16(f3.x, f3.y); r.b.w = pack2_bf16(f3.z, f3.w);
  return r;
}

// LDS tile layout: [128 rows][16 dwords] = row-major 32 bf16 K per row.
__device__ __forceinline__ void store_lds(unsigned int* sT, int tid, const Frag8& f) {
  uint4* d = (uint4*)(sT + (tid >> 1) * 16 + (tid & 1) * 8);
  d[0] = f.a; d[1] = f.b;
}

// A fragment (16x32 bf16), ISA 7.12.2: lanes 0-15 m=l, K {0..7,16..23};
// lanes 16-31 m=l-16, K {8..15,24..31}.
__device__ __forceinline__ v16bf frag_a(const unsigned int* sA, int lane, int mFrag) {
  const int m  = mFrag + (lane & 15);
  const int kh = lane >> 4;
  union { uint4 u[2]; v16bf v; } r;
  const uint4* p = (const uint4*)(sA + m * 16);
  r.u[0] = p[kh];
  r.u[1] = p[2 + kh];
  return r.v;
}

// B fragment (32x16 bf16): lane n=l&15; lanes 0-15 K {0..7,16..23}, lanes 16-31 K {8..15,24..31}.
__device__ __forceinline__ v16bf frag_b(const unsigned int* sB, int lane, int nFrag) {
  const int n  = nFrag + (lane & 15);
  const int ks = lane >> 4;
  union { uint4 u[2]; v16bf v; } r;
  const uint4* p = (const uint4*)(sB + n * 16);
  r.u[0] = p[ks];
  r.u[1] = p[2 + ks];
  return r.v;
}

// TDM: load a [tile_rows x tile_cols] bf16 tile (row stride = row_stride elems)
// from global into LDS at byte offset lds_byte_off. D# per ISA 8.3/8.4,
// groups 2/3 zero (2D tensor). This toolchain's 6-arg builtin form.
__device__ __forceinline__ void tdm_load_tile_bf16(
    const unsigned short* gsrc, unsigned int lds_byte_off,
    unsigned int tile_rows, unsigned int tile_cols,
    unsigned int tensor_rows, unsigned int tensor_cols,
    unsigned int row_stride) {
  unsigned long long ga = (unsigned long long)(uintptr_t)gsrc;
  u32x4 g0;
  g0[0] = 1u;                                   // count=1: valid user descriptor
  g0[1] = lds_byte_off;                         // lds_addr (bytes)
  g0[2] = (unsigned int)ga;                     // global_addr[31:0]
  g0[3] = (unsigned int)((ga >> 32) & 0x01FFFFFFull) | (2u << 30);  // [56:32], type=2
  i32x8 g1;
  g1[0] = (int)(1u << 16);                      // wg_mask=0, data_size=1 (2B)
  g1[1] = (int)((tensor_cols & 0xFFFFu) << 16); // tensor_dim0[15:0]
  g1[2] = (int)(((tensor_cols >> 16) & 0xFFFFu) |
                ((tensor_rows & 0xFFFFu) << 16));            // dim0[31:16], dim1[15:0]
  g1[3] = (int)(((tensor_rows >> 16) & 0xFFFFu) |
                ((tile_cols & 0xFFFFu) << 16));              // dim1[31:16], tile_dim0
  g1[4] = (int)(tile_rows & 0xFFFFu);           // tile_dim1 (tile_dim2 = 0)
  g1[5] = (int)row_stride;                      // tensor_dim0_stride[31:0]
  g1[6] = 0;                                    // stride[47:32]=0, dim1_stride=0
  g1[7] = 0;
  i32x4 gz4 = {0, 0, 0, 0};
  i32x8 gz8 = {0, 0, 0, 0, 0, 0, 0, 0};
  __builtin_amdgcn_tensor_load_to_lds(g0, g1, gz4, gz4, gz8, 0);
}

// ---------------------------------------------------------------------------
// Layer-1 GEMM: C = A(MxK,f32) * B(NxK,f32)^T, on-the-fly bf16 conversion.
// Block 128x128, BK=32, 8 waves (2x4), wave tile 64x32 -> 8 WMMA per K step.
// ---------------------------------------------------------------------------
__global__ __launch_bounds__(256) void wmma_gemm_f32_kernel(
    const float* __restrict__ A, const float* __restrict__ Bw, float* __restrict__ C,
    size_t lda, size_t ldb, size_t ldc, int kIters) {
  __shared__ unsigned int sA[128 * 16];
  __shared__ unsigned int sB[128 * 16];

  const int tid   = threadIdx.x;
  const int lane  = tid & 31;
  const int wave  = tid >> 5;
  const int waveM = wave >> 2;
  const int waveN = wave & 3;

  const float* Ap = A  + (size_t)blockIdx.y * 128 * lda;
  const float* Bp = Bw + (size_t)blockIdx.x * 128 * ldb;

  v8f zero = {};
  v8f acc[4][2];
#pragma unroll
  for (int i = 0; i < 4; ++i)
#pragma unroll
    for (int j = 0; j < 2; ++j) acc[i][j] = zero;

  Frag8 ra = fetch_tile_f32(Ap, lda, tid);
  Frag8 rb = fetch_tile_f32(Bp, ldb, tid);

  for (int kt = 0; kt < kIters; ++kt) {
    store_lds(sA, tid, ra);
    store_lds(sB, tid, rb);
    __syncthreads();

    if (kt + 1 < kIters) {
      ra = fetch_tile_f32(Ap + (size_t)(kt + 1) * 32, lda, tid);
      rb = fetch_tile_f32(Bp + (size_t)(kt + 1) * 32, ldb, tid);
    }
    if (kt + 4 < kIters) {
      __builtin_prefetch(Ap + (size_t)(tid >> 1) * lda + (size_t)(kt + 4) * 32, 0, 1);
      __builtin_prefetch(Bp + (size_t)(tid >> 1) * ldb + (size_t)(kt + 4) * 32, 0, 1);
    }

    v16bf af[4];
    v16bf bfm[2];
#pragma unroll
    for (int mi = 0; mi < 4; ++mi) af[mi] = frag_a(sA, lane, waveM * 64 + mi * 16);
#pragma unroll
    for (int ni = 0; ni < 2; ++ni) bfm[ni] = frag_b(sB, lane, waveN * 32 + ni * 16);

#pragma unroll
    for (int mi = 0; mi < 4; ++mi)
#pragma unroll
      for (int ni = 0; ni < 2; ++ni)
        acc[mi][ni] = __builtin_amdgcn_wmma_f32_16x16x32_bf16(
            false, af[mi], false, bfm[ni], (short)0, acc[mi][ni], false, false);

    __syncthreads();
  }

  const size_t mBase = (size_t)blockIdx.y * 128 + waveM * 64 + (lane >> 4) * 8;
  const size_t nBase = (size_t)blockIdx.x * 128 + waveN * 32 + (lane & 15);
#pragma unroll
  for (int mi = 0; mi < 4; ++mi)
#pragma unroll
    for (int ni = 0; ni < 2; ++ni) {
      float* outp = C + (mBase + (size_t)mi * 16) * ldc + nBase + ni * 16;
#pragma unroll
      for (int r = 0; r < 8; ++r) outp[(size_t)r * ldc] = acc[mi][ni][r];
    }
}

// ---------------------------------------------------------------------------
// Layer-2 GEMM: A is bf16 spikes -> TDM streams A tiles global->LDS (double
// buffered, TENSORcnt-synchronized); B (f32 W2) converts through registers.
// ---------------------------------------------------------------------------
__global__ __launch_bounds__(256) void wmma_gemm_bf16a_tdm_kernel(
    const unsigned short* __restrict__ A, const float* __restrict__ Bw,
    float* __restrict__ C, unsigned int lda, size_t ldb, size_t ldc, int kIters,
    unsigned int Mtot) {
  // Single static LDS block (assumed at LDS offset 0): [sA0 | sA1 | sB]
  __shared__ unsigned int smem[3 * 128 * 16];
  unsigned int* sA0 = smem;
  unsigned int* sA1 = smem + 128 * 16;
  unsigned int* sB  = smem + 2 * 128 * 16;

  const int tid   = threadIdx.x;
  const int lane  = tid & 31;
  const int wave  = tid >> 5;
  const int waveM = wave >> 2;
  const int waveN = wave & 3;

  const unsigned short* Ap = A + (size_t)blockIdx.y * 128 * lda;
  const float*          Bp = Bw + (size_t)blockIdx.x * 128 * ldb;

  v8f zero = {};
  v8f acc[4][2];
#pragma unroll
  for (int i = 0; i < 4; ++i)
#pragma unroll
    for (int j = 0; j < 2; ++j) acc[i][j] = zero;

  if (wave == 0)  // one TDM op per workgroup; TDM ignores EXEC
    tdm_load_tile_bf16(Ap, 0u, 128, 32, Mtot, lda, lda);
  Frag8 rb = fetch_tile_f32(Bp, ldb, tid);

  for (int kt = 0; kt < kIters; ++kt) {
    const unsigned int* sAcur = (kt & 1) ? sA1 : sA0;
    store_lds(sB, tid, rb);
    if (wave == 0) __builtin_amdgcn_s_wait_tensorcnt(0);  // A tile kt landed
    __syncthreads();

    if (kt + 1 < kIters) {
      if (wave == 0)
        tdm_load_tile_bf16(Ap + (size_t)(kt + 1) * 32,
                           (kt & 1) ? 0u : 8192u,  // write the other buffer
                           128, 32, Mtot, lda, lda);
      rb = fetch_tile_f32(Bp + (size_t)(kt + 1) * 32, ldb, tid);
    }

    v16bf af[4];
    v16bf bfm[2];
#pragma unroll
    for (int mi = 0; mi < 4; ++mi) af[mi] = frag_a(sAcur, lane, waveM * 64 + mi * 16);
#pragma unroll
    for (int ni = 0; ni < 2; ++ni) bfm[ni] = frag_b(sB, lane, waveN * 32 + ni * 16);

#pragma unroll
    for (int mi = 0; mi < 4; ++mi)
#pragma unroll
      for (int ni = 0; ni < 2; ++ni)
        acc[mi][ni] = __builtin_amdgcn_wmma_f32_16x16x32_bf16(
            false, af[mi], false, bfm[ni], (short)0, acc[mi][ni], false, false);

    __syncthreads();
  }

  const size_t mBase = (size_t)blockIdx.y * 128 + waveM * 64 + (lane >> 4) * 8;
  const size_t nBase = (size_t)blockIdx.x * 128 + waveN * 32 + (lane & 15);
#pragma unroll
  for (int mi = 0; mi < 4; ++mi)
#pragma unroll
    for (int ni = 0; ni < 2; ++ni) {
      float* outp = C + (mBase + (size_t)mi * 16) * ldc + nBase + ni * 16;
#pragma unroll
      for (int r = 0; r < 8; ++r) outp[(size_t)r * ldc] = acc[mi][ni][r];
    }
}

// LIF scan over T=16: v=(v+x)/2; spike = v>=1; hard reset. Spikes -> bf16.
__global__ void lif_to_bf16_kernel(const float* __restrict__ h,
                                   unsigned short* __restrict__ s, int bn_total) {
  int idx = blockIdx.x * blockDim.x + threadIdx.x;
  if (idx >= bn_total) return;
  float v = 0.0f;
#pragma unroll
  for (int t = 0; t < 16; ++t) {
    float x = h[(size_t)t * bn_total + idx];
    v = 0.5f * (v + x);
    bool spike = (v >= 1.0f);
    s[(size_t)t * bn_total + idx] = spike ? (unsigned short)0x3F80u : (unsigned short)0u;
    v = spike ? 0.0f : v;
  }
}

// Final LIF: spikes -> f32 output
__global__ void lif_to_f32_kernel(const float* __restrict__ h,
                                  float* __restrict__ s, int bn_total) {
  int idx = blockIdx.x * blockDim.x + threadIdx.x;
  if (idx >= bn_total) return;
  float v = 0.0f;
#pragma unroll
  for (int t = 0; t < 16; ++t) {
    float x = h[(size_t)t * bn_total + idx];
    v = 0.5f * (v + x);
    bool spike = (v >= 1.0f);
    s[(size_t)t * bn_total + idx] = spike ? 1.0f : 0.0f;
    v = spike ? 0.0f : v;
  }
}

// Layer 3: [512 x 1024] (bf16 spikes) x [2 x 1024]^T -> [512 x 2]; tiny.
__global__ __launch_bounds__(256) void small_gemm3_kernel(
    const unsigned short* __restrict__ s2, const float* __restrict__ W3,
    float* __restrict__ h3) {
  int tid = blockIdx.x * blockDim.x + threadIdx.x;  // 0..1023
  int m = tid >> 1;
  int o = tid & 1;
  const unsigned short* a = s2 + (size_t)m * 1024;
  const float* w = W3 + (size_t)o * 1024;
  float acc = 0.0f;
  for (int k = 0; k < 1024; ++k)
    acc += __uint_as_float((unsigned int)a[k] << 16) * w[k];
  h3[m * 2 + o] = acc;
}

extern "C" void kernel_launch(void* const* d_in, const int* in_sizes, int n_in,
                              void* d_out, int out_size, void* d_ws, size_t ws_size,
                              hipStream_t stream) {
  (void)in_sizes; (void)n_in; (void)out_size; (void)ws_size;
  const float* x  = (const float*)d_in[0];  // [16,32,60000] = [512,60000]
  const float* W1 = (const float*)d_in[1];  // [4096,60000]
  const float* W2 = (const float*)d_in[2];  // [1024,4096]
  const float* W3 = (const float*)d_in[3];  // [2,1024]
  float* out = (float*)d_out;               // [16,32,2]

  char* ws = (char*)d_ws;                                     // ~15 MB used
  float*          h1 = (float*)(ws);                          //  8 MB
  unsigned short* s1 = (unsigned short*)(ws + (8u  << 20));   //  4 MB
  float*          h2 = (float*)(ws + (12u << 20));            //  2 MB
  unsigned short* s2 = (unsigned short*)(ws + (14u << 20));   //  1 MB
  float*          h3 = (float*)(ws + (15u << 20));            //  4 KB

  // Layer 1: 512 x 60000 x 4096 (dominant) — bf16 WMMA, f32 accumulate
  wmma_gemm_f32_kernel<<<dim3(4096 / 128, 512 / 128), 256, 0, stream>>>(
      x, W1, h1, (size_t)60000, (size_t)60000, (size_t)4096, 60000 / 32);
  lif_to_bf16_kernel<<<(32 * 4096) / 256, 256, 0, stream>>>(h1, s1, 32 * 4096);

  // Layer 2: 512 x 4096 x 1024 — A spikes via Tensor Data Mover
  wmma_gemm_bf16a_tdm_kernel<<<dim3(1024 / 128, 512 / 128), 256, 0, stream>>>(
      s1, W2, h2, 4096u, (size_t)4096, (size_t)1024, 4096 / 32, 512u);
  lif_to_bf16_kernel<<<(32 * 1024) / 256, 256, 0, stream>>>(h2, s2, 32 * 1024);

  // Layer 3: 512 x 1024 x 2 — trivial
  small_gemm3_kernel<<<4, 256, 0, stream>>>(s2, W3, h3);
  lif_to_f32_kernel<<<1, 64, 0, stream>>>(h3, out, 32 * 2);
}
